// DecoderPolicyGradient_72121090834887
// MI455X (gfx1250) — compile-verified, matrix operand order checked by reading, not run on hardware
//
#include <hip/hip_runtime.h>
#include <hip/hip_bf16.h>
#include <math.h>

// Problem constants (match reference)
#define BATCH  128
#define TMAX   20
#define EDIM   512
#define HDIM   512
#define VDIM   32000

typedef __attribute__((ext_vector_type(16))) _Float16 v16h;
typedef __attribute__((ext_vector_type(8)))  float    v8f;

#define WMMA_F16(a, b, c) \
    __builtin_amdgcn_wmma_f32_16x16x32_f16(false, (a), false, (b), (short)0, (c), false, false)

// CDNA5 WMMA f16 A/B per-lane K mapping (ISA 7.12.2, 16-bit A 16x32):
// lane = half*16 + row ; pair j: vg<4 -> k=2vg+hl+8*half ; vg>=4 -> 16+2(vg-4)+hl+8*half
__device__ __forceinline__ int klocal(int j, int half) {
    int vg = j >> 1, hl = j & 1;
    int base = (vg < 4) ? (2 * vg) : (16 + 2 * (vg - 4));
    return base + hl + 8 * half;
}

__device__ __forceinline__ float sigmoidf_(float x) {
    return 1.0f / (1.0f + __expf(-x));
}

// ---------------------------------------------------------------------------
// Pack row-major [N,K] f32 (used as B = W^T) into WMMA-B tiles
// [ntile][ktile][lane][16 halfs] (contiguous 32B per lane).  Used for lin_w.
// ---------------------------------------------------------------------------
__global__ __launch_bounds__(256) void pack_wmma_b(const float* __restrict__ src,
                                                   _Float16* __restrict__ dst,
                                                   int N, int K) {
    int wave = (blockIdx.x * blockDim.x + threadIdx.x) >> 5;
    int lane = threadIdx.x & 31;
    int ktiles = K >> 5;
    int ntiles = N >> 4;
    if (wave >= ntiles * ktiles) return;
    int ntile = wave / ktiles;
    int kt    = wave % ktiles;
    int n    = ntile * 16 + (lane & 15);
    int half = lane >> 4;
    v16h v;
#pragma unroll
    for (int j = 0; j < 16; ++j) {
        int k = kt * 32 + klocal(j, half);
        v[j] = (_Float16)src[(size_t)n * K + k];
    }
    *(v16h*)(dst + ((size_t)(ntile * ktiles + kt) * 32 + lane) * 16) = v;
}

// ---------------------------------------------------------------------------
// Pack gate weights [4H, 512] f32 into gate-interleaved WMMA-B tiles:
// [jtile 0..31][kt 0..15][gate 0..3][lane][16 halfs]
// -> for a fixed jtile all tiles of the inner LSTM loop live at immediate
//    offsets (gate stride 1KB, kt stride 4KB) from one base pointer.
// ---------------------------------------------------------------------------
__global__ __launch_bounds__(256) void pack_gate_weights(const float* __restrict__ src,
                                                         _Float16* __restrict__ dst) {
    int wave = (blockIdx.x * blockDim.x + threadIdx.x) >> 5;
    int lane = threadIdx.x & 31;
    if (wave >= 32 * 16 * 4) return;
    int jtile = wave >> 6;            // 0..31
    int kt    = (wave >> 2) & 15;     // 0..15
    int gate  = wave & 3;             // 0..3
    int n    = gate * HDIM + jtile * 16 + (lane & 15);
    int half = lane >> 4;
    v16h v;
#pragma unroll
    for (int j = 0; j < 16; ++j) {
        int k = kt * 32 + klocal(j, half);
        v[j] = (_Float16)src[(size_t)n * 512 + k];
    }
    *(v16h*)(dst + ((size_t)((jtile * 16 + kt) * 4 + gate) * 32 + lane) * 16) = v;
}

// ---------------------------------------------------------------------------
// Build packed A tiles for x[t]: t==0 -> features, else embed_w[captions[b][t-1]].
// Layout: [t][btile][ktile][lane][16 halfs].
// ---------------------------------------------------------------------------
__global__ __launch_bounds__(256) void pack_x(const float* __restrict__ features,
                                              const int* __restrict__ captions,
                                              const float* __restrict__ embed_w,
                                              _Float16* __restrict__ px) {
    const int KT = EDIM / 32;             // 16
    const int BT = BATCH / 16;            // 8
    int wave = (blockIdx.x * blockDim.x + threadIdx.x) >> 5;
    int lane = threadIdx.x & 31;
    if (wave >= TMAX * BT * KT) return;
    int t     = wave / (BT * KT);
    int rem   = wave % (BT * KT);
    int btile = rem / KT;
    int kt    = rem % KT;
    int b    = btile * 16 + (lane & 15);
    int half = lane >> 4;
    const float* row;
    if (t == 0) row = features + (size_t)b * EDIM;
    else        row = embed_w + (size_t)captions[b * TMAX + (t - 1)] * EDIM;
    v16h v;
#pragma unroll
    for (int j = 0; j < 16; ++j)
        v[j] = (_Float16)row[kt * 32 + klocal(j, half)];
    size_t tile = ((size_t)t * BT + btile) * KT + kt;
    *(v16h*)(px + (tile * 32 + lane) * 16) = v;
}

// Pack h0 [B,H] f32 into WMMA-A layout [btile][ktile][lane][16]
__global__ __launch_bounds__(256) void pack_h0(const float* __restrict__ h0,
                                               _Float16* __restrict__ ph) {
    const int KT = HDIM / 32;             // 16
    const int BT = BATCH / 16;            // 8
    int wave = (blockIdx.x * blockDim.x + threadIdx.x) >> 5;
    int lane = threadIdx.x & 31;
    if (wave >= BT * KT) return;
    int btile = wave / KT;
    int kt    = wave % KT;
    int b    = btile * 16 + (lane & 15);
    int half = lane >> 4;
    v16h v;
#pragma unroll
    for (int j = 0; j < 16; ++j)
        v[j] = (_Float16)h0[(size_t)b * HDIM + kt * 32 + klocal(j, half)];
    *(v16h*)(ph + ((size_t)(btile * KT + kt) * 32 + lane) * 16) = v;
}

__global__ void copy_f32(const float* __restrict__ src, float* __restrict__ dst, int n) {
    int i = blockIdx.x * blockDim.x + threadIdx.x;
    if (i < n) dst[i] = src[i];
}

// Packed output row offsets per time step (pack_padded_sequence semantics)
__global__ void compute_toff(const int* __restrict__ lengths, int* __restrict__ toff) {
    if (blockIdx.x == 0 && threadIdx.x == 0) {
        int acc = 0;
        for (int t = 0; t < TMAX; ++t) {
            toff[t] = acc;
            int cnt = 0;
            for (int b = 0; b < BATCH; ++b) cnt += (lengths[b] > t) ? 1 : 0;
            acc += cnt;
        }
        toff[TMAX] = acc;
    }
}

// ---------------------------------------------------------------------------
// One LSTM step.  Each wave owns one (btile, jtile) 16x16 tile of h/c and
// accumulates the i,f,g,o gate tiles with a shared A operand per k-tile.
// Inner loops are register double-buffered: loads of kt+1 issue before the
// WMMAs of kt, and all tile addresses are saddr+immediate (gate-interleaved
// layout).  grid = 8 btiles * 4 jgroups, block = 256 (8 waves).
// ---------------------------------------------------------------------------
__global__ __launch_bounds__(256) void lstm_step(
    const _Float16* __restrict__ px,    // [T][8][16][512]
    const _Float16* __restrict__ pwih,  // [32 jtile][16 kt][4 gate][512]
    const _Float16* __restrict__ pwhh,  // [32 jtile][16 kt][4 gate][512]
    const float* __restrict__ b_ih, const float* __restrict__ b_hh,
    const _Float16* __restrict__ ph_in,   // [8][16][512]  (h_{t-1}, packed A)
    _Float16* __restrict__ ph_out,        // [8][16][512]  (h_t, packed A)
    _Float16* __restrict__ phs,           // [T][8][16][512]
    float* __restrict__ cbuf,             // [B][H]
    int t)
{
    __shared__ _Float16 htile[16 * 128];
    const int KT = 16;                      // 512/32
    int wv   = threadIdx.x >> 5;            // 0..7
    int lane = threadIdx.x & 31;
    int btile = blockIdx.x >> 2;            // 0..7
    int jgrp  = blockIdx.x & 3;             // 0..3
    int jtile = jgrp * 8 + wv;              // 0..31

    v8f ai = {}, af = {}, ag = {}, ao = {};

    // ---- gates += x_t @ w_ih^T  (pipelined) ----
    {
        const _Float16* ab = px + (((size_t)t * 8 + btile) * KT) * 512 + lane * 16;
        const _Float16* wb = pwih + (size_t)jtile * (16 * 4 * 512) + lane * 16;
        v16h a_c  = *(const v16h*)(ab);
        v16h b0c  = *(const v16h*)(wb + 0 * 512);
        v16h b1c  = *(const v16h*)(wb + 1 * 512);
        v16h b2c  = *(const v16h*)(wb + 2 * 512);
        v16h b3c  = *(const v16h*)(wb + 3 * 512);
#pragma unroll
        for (int kt = 0; kt < KT; ++kt) {
            v16h a_n, b0n, b1n, b2n, b3n;
            if (kt < KT - 1) {
                const _Float16* w = wb + (size_t)(kt + 1) * 2048;
                a_n = *(const v16h*)(ab + (size_t)(kt + 1) * 512);
                b0n = *(const v16h*)(w + 0 * 512);
                b1n = *(const v16h*)(w + 1 * 512);
                b2n = *(const v16h*)(w + 2 * 512);
                b3n = *(const v16h*)(w + 3 * 512);
            }
            ai = WMMA_F16(a_c, b0c, ai);
            af = WMMA_F16(a_c, b1c, af);
            ag = WMMA_F16(a_c, b2c, ag);
            ao = WMMA_F16(a_c, b3c, ao);
            if (kt < KT - 1) { a_c = a_n; b0c = b0n; b1c = b1n; b2c = b2n; b3c = b3n; }
        }
    }
    // ---- gates += h_{t-1} @ w_hh^T  (pipelined) ----
    {
        const _Float16* ab = ph_in + ((size_t)btile * KT) * 512 + lane * 16;
        const _Float16* wb = pwhh + (size_t)jtile * (16 * 4 * 512) + lane * 16;
        v16h a_c  = *(const v16h*)(ab);
        v16h b0c  = *(const v16h*)(wb + 0 * 512);
        v16h b1c  = *(const v16h*)(wb + 1 * 512);
        v16h b2c  = *(const v16h*)(wb + 2 * 512);
        v16h b3c  = *(const v16h*)(wb + 3 * 512);
#pragma unroll
        for (int kt = 0; kt < KT; ++kt) {
            v16h a_n, b0n, b1n, b2n, b3n;
            if (kt < KT - 1) {
                const _Float16* w = wb + (size_t)(kt + 1) * 2048;
                a_n = *(const v16h*)(ab + (size_t)(kt + 1) * 512);
                b0n = *(const v16h*)(w + 0 * 512);
                b1n = *(const v16h*)(w + 1 * 512);
                b2n = *(const v16h*)(w + 2 * 512);
                b3n = *(const v16h*)(w + 3 * 512);
            }
            ai = WMMA_F16(a_c, b0c, ai);
            af = WMMA_F16(a_c, b1c, af);
            ag = WMMA_F16(a_c, b2c, ag);
            ao = WMMA_F16(a_c, b3c, ao);
            if (kt < KT - 1) { a_c = a_n; b0c = b0n; b1c = b1n; b2c = b2n; b3c = b3n; }
        }
    }

    // Elementwise LSTM cell on the D-matrix layout: lane n=lane%16, VGPR r -> M=r+8*half
    int n = lane & 15, half = lane >> 4;
    int j = jtile * 16 + n;
    float bi = b_ih[0 * HDIM + j] + b_hh[0 * HDIM + j];
    float bf = b_ih[1 * HDIM + j] + b_hh[1 * HDIM + j];
    float bg = b_ih[2 * HDIM + j] + b_hh[2 * HDIM + j];
    float bo = b_ih[3 * HDIM + j] + b_hh[3 * HDIM + j];
#pragma unroll
    for (int r = 0; r < 8; ++r) {
        int m = r + 8 * half;
        int b = btile * 16 + m;
        float c_old = cbuf[(size_t)b * HDIM + j];
        float iv = sigmoidf_(ai[r] + bi);
        float fv = sigmoidf_(af[r] + bf);
        float gv = tanhf(ag[r] + bg);
        float ov = sigmoidf_(ao[r] + bo);
        float cn = fv * c_old + iv * gv;
        cbuf[(size_t)b * HDIM + j] = cn;
        float hv = ov * tanhf(cn);
        htile[m * 128 + wv * 16 + n] = (_Float16)hv;   // block's 128 local h-columns
    }
    __syncthreads();

    // Repack the block's 16x128 h slab into WMMA-A layout (4 k-tiles of 32 cols)
    if (wv < 4) {
        int ktl   = wv;
        int ktile = jgrp * 4 + ktl;                    // global h k-tile
        int m2 = lane & 15, half2 = lane >> 4;
        v16h v;
#pragma unroll
        for (int j2 = 0; j2 < 16; ++j2) {
            int kl = klocal(j2, half2);                // 0..31 within k-tile
            v[j2] = htile[m2 * 128 + ktl * 32 + kl];
        }
        size_t off = (((size_t)btile * KT + ktile) * 32 + lane) * 16;
        *(v16h*)(ph_out + off) = v;
        *(v16h*)(phs + (size_t)t * 8 * KT * 512 + off) = v;
    }
}

// ---------------------------------------------------------------------------
// Packed output projection: out[toff[t]+b, :] = hs[t,b,:] @ lin_w^T + lin_b
// Each wave computes TWO adjacent v-tiles (shared A operand, 2 independent
// WMMA chains); pipelined loads; B panel of 2nd tile at +16KB immediate.
// grid = (125, 8 btiles, 20 t), block = 256 (8 waves -> 16 v-tiles).
// ---------------------------------------------------------------------------
__global__ __launch_bounds__(256) void proj_pack(
    const _Float16* __restrict__ phs,   // [20][8][16][512]
    const _Float16* __restrict__ plin,  // [2000 ntile][16 kt][512]
    const float* __restrict__ lin_b,
    const int* __restrict__ lengths,
    const int* __restrict__ toff,
    float* __restrict__ out)
{
    const int KT = 16;                    // 512/32
    int wv   = threadIdx.x >> 5;
    int lane = threadIdx.x & 31;
    int vt0  = (blockIdx.x * 8 + wv) * 2; // even v-tile of the pair, 0..1998
    int btile = blockIdx.y;               // 0..7
    int t     = blockIdx.z;               // 0..19
    if (lengths[btile * 16] <= t) return; // lengths descending -> whole tile dropped

    v8f acc0 = {}, acc1 = {};
    const _Float16* ab = phs + (((size_t)t * 8 + btile) * KT) * 512 + lane * 16;
    const _Float16* bb = plin + ((size_t)vt0 * KT) * 512 + lane * 16;

    v16h a_c  = *(const v16h*)(ab);
    v16h b0c  = *(const v16h*)(bb);
    v16h b1c  = *(const v16h*)(bb + 16 * 512);          // +16KB: vt0+1 panel
#pragma unroll
    for (int kt = 0; kt < KT; ++kt) {
        __builtin_prefetch(bb + (size_t)(kt + 2) * 512, 0, 0);            // global_prefetch_b8
        __builtin_prefetch(bb + (size_t)(16 + kt + 2) * 512, 0, 0);
        v16h a_n, b0n, b1n;
        if (kt < KT - 1) {
            a_n = *(const v16h*)(ab + (size_t)(kt + 1) * 512);
            b0n = *(const v16h*)(bb + (size_t)(kt + 1) * 512);
            b1n = *(const v16h*)(bb + (size_t)(16 + kt + 1) * 512);
        }
        acc0 = WMMA_F16(a_c, b0c, acc0);
        acc1 = WMMA_F16(a_c, b1c, acc1);
        if (kt < KT - 1) { a_c = a_n; b0c = b0n; b1c = b1n; }
    }

    int n = lane & 15, half = lane >> 4;
    int v0 = vt0 * 16 + n;
    float bias0 = lin_b[v0];
    float bias1 = lin_b[v0 + 16];
    int base = toff[t];
#pragma unroll
    for (int r = 0; r < 8; ++r) {
        int m = r + 8 * half;
        int b = btile * 16 + m;
        if (lengths[b] > t) {
            size_t row = (size_t)(base + b) * VDIM;
            out[row + v0]      = acc0[r] + bias0;
            out[row + v0 + 16] = acc1[r] + bias1;
        }
    }
}

// ---------------------------------------------------------------------------
extern "C" void kernel_launch(void* const* d_in, const int* in_sizes, int n_in,
                              void* d_out, int out_size, void* d_ws, size_t ws_size,
                              hipStream_t stream) {
    (void)in_sizes; (void)n_in; (void)out_size; (void)ws_size;
    const float* features = (const float*)d_in[0];
    const int*   captions = (const int*)d_in[1];
    const int*   lengths  = (const int*)d_in[2];
    const float* h0       = (const float*)d_in[3];
    const float* c0       = (const float*)d_in[4];
    const float* embed_w  = (const float*)d_in[5];
    const float* w_ih     = (const float*)d_in[6];
    const float* w_hh     = (const float*)d_in[7];
    const float* b_ih     = (const float*)d_in[8];
    const float* b_hh     = (const float*)d_in[9];
    const float* lin_w    = (const float*)d_in[10];
    const float* lin_b    = (const float*)d_in[11];
    float* out = (float*)d_out;

    // Workspace carve-up (all offsets 256B-aligned)
    char* ws = (char*)d_ws;
    _Float16* pwih = (_Float16*)(ws + 0);                      //  2 MB  (32x16x4x512)
    _Float16* pwhh = (_Float16*)(ws + 2097152);                //  2 MB
    _Float16* plin = (_Float16*)(ws + 4194304);                // 32.77 MB (32000x512)
    _Float16* px   = (_Float16*)(ws + 36962304);               //  2.62 MB
    _Float16* phs  = (_Float16*)(ws + 39583744);               //  2.62 MB
    _Float16* ph0b = (_Float16*)(ws + 42205184);               //  128 KB (h ping)
    _Float16* ph1b = (_Float16*)(ws + 42336256);               //  128 KB (h pong)
    float*    cbuf = (float*)   (ws + 42467328);               //  256 KB
    int*      toff = (int*)     (ws + 42729472);               //  21 ints

    // One-time repacks (inside the graph; deterministic)
    pack_gate_weights<<<(32 * 16 * 4 * 32 + 255) / 256, 256, 0, stream>>>(w_ih, pwih);
    pack_gate_weights<<<(32 * 16 * 4 * 32 + 255) / 256, 256, 0, stream>>>(w_hh, pwhh);
    pack_wmma_b<<<(32000 * 16 * 32 + 255) / 256, 256, 0, stream>>>(lin_w, plin, VDIM, HDIM);
    pack_x<<<(TMAX * 8 * 16 * 32 + 255) / 256, 256, 0, stream>>>(features, captions, embed_w, px);
    pack_h0<<<(8 * 16 * 32 + 255) / 256, 256, 0, stream>>>(h0, ph0b);
    copy_f32<<<(BATCH * HDIM + 255) / 256, 256, 0, stream>>>(c0, cbuf, BATCH * HDIM);
    compute_toff<<<1, 32, 0, stream>>>(lengths, toff);

    // Recurrence: 20 sequential step launches, double-buffered h
    for (int t = 0; t < TMAX; ++t) {
        const _Float16* hin  = (t & 1) ? ph1b : ph0b;
        _Float16*       hout = (t & 1) ? ph0b : ph1b;
        lstm_step<<<32, 256, 0, stream>>>(px, pwih, pwhh, b_ih, b_hh,
                                          hin, hout, phs, cbuf, t);
    }

    // Packed projection: writes every output row exactly once
    proj_pack<<<dim3(VDIM / 16 / 16, 8, TMAX), 256, 0, stream>>>(phs, plin, lin_b,
                                                                 lengths, toff, out);
}
